// NeuralBlochRK4_21191368638759
// MI455X (gfx1250) — compile-verified, matrix operand order checked by reading, not run on hardware
//
#include <hip/hip_runtime.h>
#include <hip/hip_bf16.h>
#include <math.h>

#define Bsz 16384
#define Tn  256
#define HID 128
#define WAVES_PER_BLOCK 2
#define BLOCK_THREADS (WAVES_PER_BLOCK * 32)
#define HST 136   // h-stage row stride in halves (128 + 8 pad -> conflict-free b128)
#define XST 40    // x-stage row stride in halves (32 + 8 pad)

typedef __attribute__((ext_vector_type(16))) _Float16 v16h;
typedef __attribute__((ext_vector_type(8)))  _Float16 v8h;
typedef __attribute__((ext_vector_type(8)))  float    v8f;

// Wave-private LDS ordering fence: stores visible before subsequent ds loads,
// and stops the compiler from reordering across it.
#define LDS_FENCE() asm volatile("s_wait_dscnt 0" ::: "memory")

// Branch-free tanh on the hardware transcendental pipe:
// tanh(x) = 1 - 2/(exp(2x)+1);  exp(2x) = exp2(x * 2*log2(e)).
// Saturates correctly: x>>0 -> exp2=+inf -> rcp=0 -> 1;  x<<0 -> exp2=0 -> -1.
static __device__ __forceinline__ float fast_tanh(float x) {
  float e = __builtin_amdgcn_exp2f(x * 2.885390081777927f);
  return 1.0f - 2.0f * __builtin_amdgcn_rcpf(e + 1.0f);
}

static __device__ __forceinline__ v16h load_frag2(const _Float16* c0, const _Float16* c1) {
  v8h a = *(const v8h*)c0;   // ds_load_b128
  v8h b = *(const v8h*)c1;   // ds_load_b128
  v16h r;
#pragma unroll
  for (int i = 0; i < 8; ++i) { r[i] = a[i]; r[i + 8] = b[i]; }
  return r;
}

static __device__ __forceinline__ v8f wmma_f16(v16h a, v16h b, v8f c) {
  return __builtin_amdgcn_wmma_f32_16x16x32_f16(false, a, false, b, (short)0, c, false, false);
}

__global__ __launch_bounds__(BLOCK_THREADS)
void node_rk4_kernel(const float* __restrict__ y0, const float* __restrict__ tg,
                     const float* __restrict__ u,  const float* __restrict__ p,
                     const float* __restrict__ W1, const float* __restrict__ b1,
                     const float* __restrict__ W2, const float* __restrict__ b2,
                     const float* __restrict__ W3, const float* __restrict__ b3,
                     float* __restrict__ out)
{
  // Weight fragments packed in B-matrix lane order: half index = c*256 + lane*8 + e,
  // maps to element (k = ks*32 + c*16 + (lane/16)*8 + e, n = nt*16 + lane%16).
  __shared__ _Float16 w1p[8][512];    //  8 KB : 8 N-tiles, K=0..31 (13 real, rest 0)
  __shared__ _Float16 w2p[32][512];   // 32 KB : 8 N-tiles x 4 K-steps
  __shared__ _Float16 w3p[4][512];    //  4 KB : 1 N-tile (3 real cols) x 4 K-steps
  __shared__ _Float16 hstage[WAVES_PER_BLOCK][16 * HST]; // per-wave activation stage
  __shared__ _Float16 xstage[WAVES_PER_BLOCK][16 * XST]; // per-wave input stage (K-padded)
  __shared__ float    kscr[WAVES_PER_BLOCK][16 * 4];     // per-wave RK4 slope scratch

  const int tid  = threadIdx.x;
  const int w    = tid >> 5;
  const int lw   = tid & 31;
  const int half = lw >> 4;
  const int nIn  = lw & 15;

  // ---- one-time: convert weights fp32 -> f16 fragments in LDS ----
  for (int idx = tid; idx < 32 * 512; idx += BLOCK_THREADS) {
    int f = idx >> 9, r = idx & 511;
    int c = r >> 8, lane = (r >> 3) & 31, e = r & 7;
    int nt = f >> 2, ks = f & 3;
    int k = ks * 32 + c * 16 + ((lane >> 4) << 3) + e;
    int n = nt * 16 + (lane & 15);
    w2p[f][r] = (_Float16)W2[n * HID + k];
  }
  for (int idx = tid; idx < 8 * 512; idx += BLOCK_THREADS) {
    int f = idx >> 9, r = idx & 511;
    int c = r >> 8, lane = (r >> 3) & 31, e = r & 7;
    int k = c * 16 + ((lane >> 4) << 3) + e;
    int n = f * 16 + (lane & 15);
    w1p[f][r] = (k < 13) ? (_Float16)W1[n * 13 + k] : (_Float16)0.0f;
  }
  for (int idx = tid; idx < 4 * 512; idx += BLOCK_THREADS) {
    int f = idx >> 9, r = idx & 511;
    int c = r >> 8, lane = (r >> 3) & 31, e = r & 7;
    int k = f * 32 + c * 16 + ((lane >> 4) << 3) + e;
    int n = lane & 15;
    w3p[f][r] = (n < 3) ? (_Float16)W3[n * HID + k] : (_Float16)0.0f;
  }

  // zero x-stage so K padding (cols 13..31) contributes 0
  for (int i = lw; i < 16 * XST; i += 32) xstage[w][i] = (_Float16)0.0f;

  const int gw = blockIdx.x * WAVES_PER_BLOCK + w;
  const int row = gw * 16 + nIn;   // valid for lanes 0..15

  // per-lane bias registers (col fixed per lane in C layout)
  float bias1[8], bias2[8];
#pragma unroll
  for (int nt = 0; nt < 8; ++nt) {
    bias1[nt] = b1[nt * 16 + nIn];
    bias2[nt] = b2[nt * 16 + nIn];
  }
  float bias3 = (nIn < 3) ? b3[nIn] : 0.0f;

  float yv[3] = {0.f, 0.f, 0.f};
  if (lw < 16) {
    yv[0] = y0[row * 3 + 0]; yv[1] = y0[row * 3 + 1]; yv[2] = y0[row * 3 + 2];
    _Float16* xr = &xstage[w][lw * XST];
#pragma unroll
    for (int j = 0; j < 5; ++j) xr[7 + j] = (_Float16)p[row * 5 + j]; // params: constant cols
    float* op = out + (size_t)row * Tn * 3;
    op[0] = yv[0]; op[1] = yv[1]; op[2] = yv[2];                      // out[:,0,:] = y0
  }
  __syncthreads();

  // u at t0 (uniform grid => interp(t_n) == u[:,n,:])
  float uc[4] = {0, 0, 0, 0};
  if (lw < 16) {
    const float4 u4 = *(const float4*)(u + ((size_t)row * Tn) * 4);
    uc[0] = u4.x; uc[1] = u4.y; uc[2] = u4.z; uc[3] = u4.w;
  }

#pragma unroll 1
  for (int n = 0; n < Tn - 1; ++n) {
    const float tn  = tg[n];
    const float tn1 = tg[n + 1];
    const float hs  = tn1 - tn;

    float un1[4] = {0, 0, 0, 0};
    if (lw < 16) {
      const float4 u4 = *(const float4*)(u + ((size_t)row * Tn + n + 1) * 4);
      un1[0] = u4.x; un1[1] = u4.y; un1[2] = u4.z; un1[3] = u4.w;
    }
    float um[4];
#pragma unroll
    for (int j = 0; j < 4; ++j) um[j] = 0.5f * (uc[j] + un1[j]);

    float kst[4][3];

#pragma unroll
    for (int s = 0; s < 4; ++s) {
      // ---- stage input selection (fp32) ----
      float ys[3], us[4], ts;
      if (s == 0) {
#pragma unroll
        for (int c = 0; c < 3; ++c) ys[c] = yv[c];
#pragma unroll
        for (int j = 0; j < 4; ++j) us[j] = uc[j];
        ts = tn;
      } else if (s == 3) {
#pragma unroll
        for (int c = 0; c < 3; ++c) ys[c] = yv[c] + hs * kst[2][c];
#pragma unroll
        for (int j = 0; j < 4; ++j) us[j] = un1[j];
        ts = tn1;
      } else {
#pragma unroll
        for (int c = 0; c < 3; ++c) ys[c] = yv[c] + 0.5f * hs * kst[s - 1][c];
#pragma unroll
        for (int j = 0; j < 4; ++j) us[j] = um[j];
        ts = tn + 0.5f * hs;
      }

      if (lw < 16) {
        _Float16* xr = &xstage[w][lw * XST];
        xr[0] = (_Float16)ys[0]; xr[1] = (_Float16)ys[1]; xr[2] = (_Float16)ys[2];
        xr[3] = (_Float16)us[0]; xr[4] = (_Float16)us[1];
        xr[5] = (_Float16)us[2]; xr[6] = (_Float16)us[3];
        xr[12] = (_Float16)ts;
      }
      LDS_FENCE();

      // ---- layer 1: (16x32) x (32x128), K padded with zeros ----
      const _Float16* xb = &xstage[w][nIn * XST + half * 8];
      v16h ax = load_frag2(xb, xb + 16);
#pragma unroll
      for (int nt = 0; nt < 8; ++nt) {
        v16h bf = load_frag2(&w1p[nt][lw * 8], &w1p[nt][256 + lw * 8]);
        v8f acc;
#pragma unroll
        for (int i = 0; i < 8; ++i) acc[i] = bias1[nt];
        acc = wmma_f16(ax, bf, acc);
#pragma unroll
        for (int v = 0; v < 8; ++v)
          hstage[w][(half * 8 + v) * HST + nt * 16 + nIn] = (_Float16)fast_tanh(acc[v]);
      }
      LDS_FENCE();

      // ---- layer 2: (16x128) x (128x128), 8 N-tiles x 4 K-steps ----
      v8f acc2[8];
#pragma unroll
      for (int nt = 0; nt < 8; ++nt)
#pragma unroll
        for (int i = 0; i < 8; ++i) acc2[nt][i] = bias2[nt];
#pragma unroll
      for (int ks = 0; ks < 4; ++ks) {
        const _Float16* ab = &hstage[w][nIn * HST + ks * 32 + half * 8];
        v16h af = load_frag2(ab, ab + 16);
#pragma unroll
        for (int nt = 0; nt < 8; ++nt) {
          v16h bf = load_frag2(&w2p[nt * 4 + ks][lw * 8], &w2p[nt * 4 + ks][256 + lw * 8]);
          acc2[nt] = wmma_f16(af, bf, acc2[nt]);
        }
      }
      // tanh + restage h2 (h1 fragment loads already landed in VGPRs; LDS is in-order per wave)
#pragma unroll
      for (int nt = 0; nt < 8; ++nt)
#pragma unroll
        for (int v = 0; v < 8; ++v)
          hstage[w][(half * 8 + v) * HST + nt * 16 + nIn] = (_Float16)fast_tanh(acc2[nt][v]);
      LDS_FENCE();

      // ---- layer 3: (16x128) x (128x16), only 3 cols real ----
      v8f acc3;
#pragma unroll
      for (int i = 0; i < 8; ++i) acc3[i] = bias3;
#pragma unroll
      for (int ks = 0; ks < 4; ++ks) {
        const _Float16* ab = &hstage[w][nIn * HST + ks * 32 + half * 8];
        v16h af = load_frag2(ab, ab + 16);
        v16h bf = load_frag2(&w3p[ks][lw * 8], &w3p[ks][256 + lw * 8]);
        acc3 = wmma_f16(af, bf, acc3);
      }
      if (nIn < 3) {
#pragma unroll
        for (int v = 0; v < 8; ++v) kscr[w][(half * 8 + v) * 4 + nIn] = acc3[v];
      }
      LDS_FENCE();
      if (lw < 16) {
        kst[s][0] = kscr[w][lw * 4 + 0];
        kst[s][1] = kscr[w][lw * 4 + 1];
        kst[s][2] = kscr[w][lw * 4 + 2];
      }
    } // RK4 stages

    if (lw < 16) {
#pragma unroll
      for (int c = 0; c < 3; ++c)
        yv[c] += (hs / 6.0f) * (kst[0][c] + 2.0f * kst[1][c] + 2.0f * kst[2][c] + kst[3][c]);
      float* op = out + ((size_t)row * Tn + n + 1) * 3;
      op[0] = yv[0]; op[1] = yv[1]; op[2] = yv[2];
    }
#pragma unroll
    for (int j = 0; j < 4; ++j) uc[j] = un1[j];
  }
}

extern "C" void kernel_launch(void* const* d_in, const int* in_sizes, int n_in,
                              void* d_out, int out_size, void* d_ws, size_t ws_size,
                              hipStream_t stream) {
  (void)in_sizes; (void)n_in; (void)d_ws; (void)ws_size; (void)out_size;
  const float* y0 = (const float*)d_in[0];
  const float* tg = (const float*)d_in[1];
  const float* u  = (const float*)d_in[2];
  const float* p  = (const float*)d_in[3];
  const float* W1 = (const float*)d_in[4];
  const float* b1 = (const float*)d_in[5];
  const float* W2 = (const float*)d_in[6];
  const float* b2 = (const float*)d_in[7];
  const float* W3 = (const float*)d_in[8];
  const float* b3 = (const float*)d_in[9];
  float* out = (float*)d_out;

  dim3 grid(Bsz / (16 * WAVES_PER_BLOCK));   // 512 blocks x 2 waves = 1024 waves
  dim3 block(BLOCK_THREADS);
  hipLaunchKernelGGL(node_rk4_kernel, grid, block, 0, stream,
                     y0, tg, u, p, W1, b1, W2, b2, W3, b3, out);
}
